// EgoCrossAttnPool_70720931496520
// MI455X (gfx1250) — matrix operand order repaired and checked by reading, not compiled, not run on hardware
//
#include <hip/hip_runtime.h>

// ---------------------------------------------------------------------------
// EgoCrossAttnPool fused kernel for gfx1250 (MI455X), bf16 WMMA path.
// B=8, N=2048, K=32, D=128, H=4, HD=32  ->  BN=16384 tokens.
// ---------------------------------------------------------------------------

typedef __attribute__((ext_vector_type(16))) __bf16 v16bf;
typedef __attribute__((ext_vector_type(8)))  float  v8f;

#define BN_TOT 16384
#define Kn     32
#define Dn     128
#define Hn     4
#define HDn    32
#define TM     8              // tokens per workgroup
#define ROWS   (TM * Kn)      // 256 kv rows per workgroup
#define LN_EPS 1e-6f

// ---------------- wave32 reductions -----------------------------------------
__device__ inline float wave_sum(float v) {
  #pragma unroll
  for (int o = 16; o > 0; o >>= 1) v += __shfl_xor(v, o, 32);
  return v;
}
__device__ inline float wave_max(float v) {
  #pragma unroll
  for (int o = 16; o > 0; o >>= 1) v = fmaxf(v, __shfl_xor(v, o, 32));
  return v;
}

// ---------------- WMMA helpers ----------------------------------------------
__device__ inline v8f wmma_bf16(v16bf a, v16bf b, v8f c) {
  // (neg_a, A, neg_b, B, c_mod, C, reuse_a, reuse_b)
  return __builtin_amdgcn_wmma_f32_16x16x32_bf16(false, a, false, b, (short)0, c,
                                                 false, false);
}

// A fragment (16x32 bf16) from row-major LDS [row][128].
// Per ISA: lane m = lane&15; lanes 0-15 hold K {0..7,16..23}, lanes 16-31
// hold K {8..15,24..31} -> two contiguous 16B chunks per lane.
__device__ inline v16bf load_A(const __bf16* s, int row0, int kt, int lane) {
  const __bf16* p = s + (row0 + (lane & 15)) * Dn + kt * 32 + ((lane >> 4) * 8);
  v16bf a;
  ((uint4*)&a)[0] = *(const uint4*)(p);
  ((uint4*)&a)[1] = *(const uint4*)(p + 16);
  return a;
}

// B fragment (32x16 bf16) from pre-swizzled global weights:
// frag(wid,kt,nt) is 512 bf16, lane-major (16 halves per lane, contiguous).
__device__ inline v16bf load_B(const __bf16* w, int wid, int kt, int nt, int lane) {
  const __bf16* p = w + (((wid * 4 + kt) * 8 + nt) * 512) + lane * 16;
  v16bf b;
  ((uint4*)&b)[0] = *(const uint4*)(p);
  ((uint4*)&b)[1] = *(const uint4*)(p + 8);
  return b;
}

// ---------------- LayerNorm one 128-wide row with one wave -------------------
__device__ inline void ln_row_to_bf16(const float* __restrict__ g, __bf16* dst,
                                      const float* __restrict__ sc,
                                      const float* __restrict__ bi, int lane) {
  float x0 = g[lane], x1 = g[lane + 32], x2 = g[lane + 64], x3 = g[lane + 96];
  float s  = wave_sum(x0 + x1 + x2 + x3);
  float sq = wave_sum(x0 * x0 + x1 * x1 + x2 * x2 + x3 * x3);
  float mu  = s * (1.0f / 128.0f);
  float var = sq * (1.0f / 128.0f) - mu * mu;
  float r   = rsqrtf(var + LN_EPS);
  dst[lane]      = (__bf16)((x0 - mu) * r * sc[lane]      + bi[lane]);
  dst[lane + 32] = (__bf16)((x1 - mu) * r * sc[lane + 32] + bi[lane + 32]);
  dst[lane + 64] = (__bf16)((x2 - mu) * r * sc[lane + 64] + bi[lane + 64]);
  dst[lane + 96] = (__bf16)((x3 - mu) * r * sc[lane + 96] + bi[lane + 96]);
}

// ---------------- prep: f32 weights -> bf16 pre-swizzled B fragments ---------
// Layout: dst[((wid*4+kt)*8+nt)*512 + lane*16 + h] = W[k][n],
//   n = nt*16 + (lane&15),  k = kt*32 + (lane>>4)*8 + (h&7) + (h>>3)*16.
__global__ void prep_weights_kernel(const float* __restrict__ Wq,
                                    const float* __restrict__ Wk,
                                    const float* __restrict__ Wv,
                                    const float* __restrict__ Wo,
                                    __bf16* __restrict__ dst) {
  int o0 = (blockIdx.x * blockDim.x + threadIdx.x) * 4;
  int w  = (o0 >> 14) & 3;  // uniform across the 4 consecutive outputs
  const float* W = (w == 0) ? Wq : (w == 1) ? Wk : (w == 2) ? Wv : Wo;
  #pragma unroll
  for (int i = 0; i < 4; ++i) {
    int o  = o0 + i;
    int h  = o & 15;
    int ln = (o >> 4) & 31;
    int nt = (o >> 9) & 7;
    int kt = (o >> 12) & 3;
    int k  = kt * 32 + (ln >> 4) * 8 + (h & 7) + (h >> 3) * 16;
    int n  = nt * 16 + (ln & 15);
    dst[o] = (__bf16)W[k * 128 + n];
  }
}

// ---------------- main fused kernel ------------------------------------------
__launch_bounds__(256)
__global__ void ego_attn_kernel(const float* __restrict__ ego_q,
                                const float* __restrict__ nbr,
                                const int*   __restrict__ mask,
                                const float* __restrict__ lnq_s,
                                const float* __restrict__ lnq_b,
                                const float* __restrict__ lnk_s,
                                const float* __restrict__ lnk_b,
                                const float* __restrict__ bq,
                                const float* __restrict__ bk,
                                const float* __restrict__ bv,
                                const float* __restrict__ bo,
                                const __bf16* __restrict__ wswz,
                                float* __restrict__ out) {
  const int tid  = threadIdx.x;
  const int lane = tid & 31;
  const int wv   = tid >> 5;                 // 8 waves
  const long t0  = (long)blockIdx.x * TM;    // first token of this WG

  __shared__ __bf16 sA[ROWS * Dn];           // LN'd kv (A matrix)      64 KB
  __shared__ __bf16 sP[ROWS * Dn];           // projected K, then V     64 KB
  __shared__ __bf16 sQ[16 * Dn];             // LN'd q, zero-padded      4 KB
  __shared__ float  sQh[16 * Dn];            // q proj, then y staging   8 KB
  __shared__ float  sAttn[TM * Hn * Kn];     // softmax weights          4 KB
  __shared__ __bf16 sOut[16 * Dn];           // attn output (A matrix)   4 KB
  __shared__ float  sHasAny[TM];

  // Get this wave's kt=0 Wk and Wv B-fragments in flight while we LayerNorm.
  __builtin_prefetch(wswz + ((1 * 4 * 8 + wv) * 512), 0, 1);
  __builtin_prefetch(wswz + ((2 * 4 * 8 + wv) * 512), 0, 1);

  // --- Stage 0: LayerNorm q + kv into bf16 LDS -------------------------------
  ln_row_to_bf16(ego_q + (t0 + wv) * Dn, sQ + wv * Dn, lnq_s, lnq_b, lane);
  for (int j = lane; j < Dn; j += 32) {      // zero pad rows 8..15
    sQ[(wv + 8) * Dn + j]   = (__bf16)0.0f;
    sOut[(wv + 8) * Dn + j] = (__bf16)0.0f;
  }
  for (int r = wv; r < ROWS; r += 8)
    ln_row_to_bf16(nbr + (t0 * Kn + r) * (long)Dn, sA + r * Dn, lnk_s, lnk_b, lane);
  __syncthreads();

  // --- Stage 1: Q projection (16x128)@(128x128), wave wv owns N-tile wv ------
  {
    v16bf b0 = load_B(wswz, 0, 0, wv, lane);
    v16bf b1 = load_B(wswz, 0, 1, wv, lane);
    v16bf b2 = load_B(wswz, 0, 2, wv, lane);
    v16bf b3 = load_B(wswz, 0, 3, wv, lane);
    v8f acc = {};
    acc = wmma_bf16(load_A(sQ, 0, 0, lane), b0, acc);
    acc = wmma_bf16(load_A(sQ, 0, 1, lane), b1, acc);
    acc = wmma_bf16(load_A(sQ, 0, 2, lane), b2, acc);
    acc = wmma_bf16(load_A(sQ, 0, 3, lane), b3, acc);
    const int   n    = wv * 16 + (lane & 15);
    const float bias = bq[n];
    const int   mb   = (lane >> 4) * 8;
    #pragma unroll
    for (int r = 0; r < 8; ++r)
      sQh[(mb + r) * Dn + n] = (acc[r] + bias) * 0.17677669529663688f; // /sqrt(32)
  }
  __syncthreads();

  // --- Stage 2: K projection (256x128)@(128x128) -> sP (bf16) ----------------
  {
    v16bf b0 = load_B(wswz, 1, 0, wv, lane);
    v16bf b1 = load_B(wswz, 1, 1, wv, lane);
    v16bf b2 = load_B(wswz, 1, 2, wv, lane);
    v16bf b3 = load_B(wswz, 1, 3, wv, lane);
    const int   n    = wv * 16 + (lane & 15);
    const float bias = bk[n];
    const int   mb   = (lane >> 4) * 8;
    #pragma unroll 4
    for (int mt = 0; mt < 16; ++mt) {
      v8f acc = {};
      acc = wmma_bf16(load_A(sA, mt * 16, 0, lane), b0, acc);
      acc = wmma_bf16(load_A(sA, mt * 16, 1, lane), b1, acc);
      acc = wmma_bf16(load_A(sA, mt * 16, 2, lane), b2, acc);
      acc = wmma_bf16(load_A(sA, mt * 16, 3, lane), b3, acc);
      #pragma unroll
      for (int r = 0; r < 8; ++r)
        sP[(mt * 16 + mb + r) * Dn + n] = (__bf16)(acc[r] + bias);
    }
  }
  __syncthreads();

  // --- Stage 3: masked scores + softmax (lane == key) ------------------------
  {
    #pragma unroll
    for (int j = 0; j < 4; ++j) {
      const int p = wv * 4 + j;          // (token, head) pair 0..31
      const int t = p >> 2, h = p & 3;
      const int   mk      = mask[(t0 + t) * Kn + lane];
      const float anyTot  = wave_sum((mk != 0) ? 1.0f : 0.0f);
      const bool  hasAny  = anyTot > 0.5f;
      const bool  msafe   = hasAny ? (mk != 0) : (lane == 0);
      const float* qrow   = sQh + t * Dn + h * HDn;
      const __bf16* krow  = sP + (t * Kn + lane) * Dn + h * HDn;
      float s = 0.0f;
      #pragma unroll
      for (int d = 0; d < 32; ++d) {
        int dd = (d + lane) & 31;        // rotate to spread LDS banks
        s += qrow[dd] * (float)krow[dd];
      }
      s = msafe ? s : -3.0e38f;
      const float mx  = wave_max(s);
      const float pr  = msafe ? __expf(s - mx) : 0.0f;
      const float sum = wave_sum(pr);
      sAttn[(t * Hn + h) * Kn + lane] = pr / sum;
      if (h == 0 && lane == 0) sHasAny[t] = hasAny ? 1.0f : 0.0f;
    }
  }
  __syncthreads();

  // --- Stage 4: V projection -> sP (reuse) -----------------------------------
  {
    v16bf b0 = load_B(wswz, 2, 0, wv, lane);
    v16bf b1 = load_B(wswz, 2, 1, wv, lane);
    v16bf b2 = load_B(wswz, 2, 2, wv, lane);
    v16bf b3 = load_B(wswz, 2, 3, wv, lane);
    const int   n    = wv * 16 + (lane & 15);
    const float bias = bv[n];
    const int   mb   = (lane >> 4) * 8;
    #pragma unroll 4
    for (int mt = 0; mt < 16; ++mt) {
      v8f acc = {};
      acc = wmma_bf16(load_A(sA, mt * 16, 0, lane), b0, acc);
      acc = wmma_bf16(load_A(sA, mt * 16, 1, lane), b1, acc);
      acc = wmma_bf16(load_A(sA, mt * 16, 2, lane), b2, acc);
      acc = wmma_bf16(load_A(sA, mt * 16, 3, lane), b3, acc);
      #pragma unroll
      for (int r = 0; r < 8; ++r)
        sP[(mt * 16 + mb + r) * Dn + n] = (__bf16)(acc[r] + bias);
    }
  }
  __syncthreads();

  // --- Stage 5: attn @ V -> sOut (bf16 A matrix for output proj) -------------
  {
    const int t  = tid >> 5;             // token
    const int d0 = (tid & 31) * 4;       // 4 consecutive feature dims
    #pragma unroll
    for (int dd = 0; dd < 4; ++dd) {
      const int d = d0 + dd;
      const int h = d >> 5;
      const float* ar = sAttn + (t * Hn + h) * Kn;
      float s = 0.0f;
      #pragma unroll
      for (int k = 0; k < 32; ++k)
        s += ar[k] * (float)sP[(t * Kn + k) * Dn + d];
      sOut[t * Dn + d] = (__bf16)s;
    }
  }
  __syncthreads();

  // --- Stage 6: output projection (16x128)@(128x128) -> sQh staging ----------
  {
    v16bf b0 = load_B(wswz, 3, 0, wv, lane);
    v16bf b1 = load_B(wswz, 3, 1, wv, lane);
    v16bf b2 = load_B(wswz, 3, 2, wv, lane);
    v16bf b3 = load_B(wswz, 3, 3, wv, lane);
    v8f acc = {};
    acc = wmma_bf16(load_A(sOut, 0, 0, lane), b0, acc);
    acc = wmma_bf16(load_A(sOut, 0, 1, lane), b1, acc);
    acc = wmma_bf16(load_A(sOut, 0, 2, lane), b2, acc);
    acc = wmma_bf16(load_A(sOut, 0, 3, lane), b3, acc);
    const int   n    = wv * 16 + (lane & 15);
    const float bias = bo[n];
    const int   mb   = (lane >> 4) * 8;
    #pragma unroll
    for (int r = 0; r < 8; ++r)
      sQh[(mb + r) * Dn + n] = acc[r] + bias;   // all 16 rows; EXEC stays full
  }
  __syncthreads();

  // --- Stage 7: masked coalesced store ---------------------------------------
  {
    const int   t  = tid >> 5;
    const float g  = sHasAny[t];
    const int   d0 = (tid & 31) * 4;
    #pragma unroll
    for (int dd = 0; dd < 4; ++dd) {
      const int d = d0 + dd;
      out[(t0 + t) * Dn + d] = sQh[t * Dn + d] * g;
    }
  }
}

// ---------------------------------------------------------------------------
extern "C" void kernel_launch(void* const* d_in, const int* in_sizes, int n_in,
                              void* d_out, int out_size, void* d_ws, size_t ws_size,
                              hipStream_t stream) {
  const float* ego_q = (const float*)d_in[0];
  const float* nbr   = (const float*)d_in[1];
  const int*   mask  = (const int*)d_in[2];
  const float* lnq_s = (const float*)d_in[3];
  const float* lnq_b = (const float*)d_in[4];
  const float* lnk_s = (const float*)d_in[5];
  const float* lnk_b = (const float*)d_in[6];
  const float* Wq    = (const float*)d_in[7];
  const float* bq    = (const float*)d_in[8];
  const float* Wk    = (const float*)d_in[9];
  const float* bk    = (const float*)d_in[10];
  const float* Wv    = (const float*)d_in[11];
  const float* bv    = (const float*)d_in[12];
  const float* Wo    = (const float*)d_in[13];
  const float* bo    = (const float*)d_in[14];

  __bf16* wswz = (__bf16*)d_ws;  // 4 weights * 16384 bf16 = 128 KB scratch

  prep_weights_kernel<<<64, 256, 0, stream>>>(Wq, Wk, Wv, Wo, wswz);
  ego_attn_kernel<<<BN_TOT / TM, 256, 0, stream>>>(
      ego_q, nbr, mask, lnq_s, lnq_b, lnk_s, lnk_b,
      bq, bk, bv, bo, wswz, (float*)d_out);
}
